// RWKVAttention_8383776161802
// MI455X (gfx1250) — compile-verified, matrix operand order checked by reading, not run on hardware
//
#include <hip/hip_runtime.h>

#define HIDDEN   2048
#define HEADS    32
#define HEAD_DIM 64
#define TLEN     2048

typedef __attribute__((ext_vector_type(16))) _Float16 v16h;
typedef __attribute__((ext_vector_type(8)))  _Float16 v8h;
typedef __attribute__((ext_vector_type(8)))  float    v8f;

// ---------------------------------------------------------------------------
// f32 -> f16 weight conversion (streamed once; keeps GEMM operands compact)
// ---------------------------------------------------------------------------
__global__ void f32_to_f16_kernel(const float* __restrict__ src,
                                  _Float16* __restrict__ dst, int n) {
    int i = blockIdx.x * blockDim.x + threadIdx.x;
    int stride = gridDim.x * blockDim.x;
    for (; i < n; i += stride) dst[i] = (_Float16)src[i];
}

// ---------------------------------------------------------------------------
// Token-shift mixing: xk/xv/xr/xg = x + (x[t-1]-x[t]) * (1 - tm_*)
// ---------------------------------------------------------------------------
__global__ void mix_kernel(const float* __restrict__ x, const int* __restrict__ pos,
                           const float* __restrict__ tm_k, const float* __restrict__ tm_v,
                           const float* __restrict__ tm_r, const float* __restrict__ tm_g,
                           _Float16* __restrict__ xk, _Float16* __restrict__ xv,
                           _Float16* __restrict__ xr, _Float16* __restrict__ xg) {
    int idx = blockIdx.x * blockDim.x + threadIdx.x;   // grid covers T*C exactly
    int t = idx >> 11;        // / HIDDEN
    int c = idx & (HIDDEN - 1);
    float xc   = x[idx];
    float prev = (pos[t] == 0) ? 0.f : x[idx - HIDDEN];
    float xx   = prev - xc;
    xk[idx] = (_Float16)(xc + xx * (1.f - tm_k[c]));
    xv[idx] = (_Float16)(xc + xx * (1.f - tm_v[c]));
    xr[idx] = (_Float16)(xc + xx * (1.f - tm_r[c]));
    xg[idx] = (_Float16)(xc + xx * (1.f - tm_g[c]));
}

// ---------------------------------------------------------------------------
// Fragment loader: two b128 loads per 16x32 (or 32x16) f16 fragment.
// ---------------------------------------------------------------------------
__device__ __forceinline__ v16h load_frag(const _Float16* __restrict__ p) {
    v8h lo = *(const v8h*)p;
    v8h hi = *(const v8h*)(p + 16);
    v16h r;
    #pragma unroll
    for (int i = 0; i < 8; ++i) { r[i] = lo[i]; r[i + 8] = hi[i]; }
    return r;
}

// ---------------------------------------------------------------------------
// Y[M,N] = A[M,K] @ W[N,K]^T via v_wmma_f32_16x16x32_f16.
// One wave -> 16x64 output strip. Unroll-by-2 ping-pong double buffer:
// buffers X/Y alternate, each iteration consumes X (k-64) and refills it
// with k, then consumes Y (k-32) and refills with k+32. No register
// rotation copies (avoids v_dual_mov churn + WMMA WAR v_nops); loads keep
// ~4 WMMAs + 10 loads of slack so waits stay at partial load counts.
// A-layout per ISA 7.12.2 (16-bit A 16x32): lane<16 holds K{0..7,16..23},
// lane>=16 holds K{8..15,24..31}; B mirrors with lanes = N columns = W rows.
// ---------------------------------------------------------------------------
__global__ __launch_bounds__(256) void wmma_gemm_f16(
        const _Float16* __restrict__ A, const _Float16* __restrict__ W,
        float* __restrict__ C, int M, int N, int K, int silu)
{
    const int lane = threadIdx.x & 31;
    const int wave = threadIdx.x >> 5;           // 0..7
    const int m0   = blockIdx.y * 128 + wave * 16;
    const int n0   = blockIdx.x * 64;
    const int row  = lane & 15;
    const int kg   = lane >> 4;                  // 0 or 1

    const _Float16* ap = A + (size_t)(m0 + row) * K + kg * 8;
    const _Float16* wp = W + (size_t)(n0 + row) * K + kg * 8;

    v8f acc[4] = {v8f{}, v8f{}, v8f{}, v8f{}};

    // ---- prologue: fill both buffers (k=0 and k=32) ----------------------
    v16h aX = load_frag(ap);
    v16h bX[4];
    #pragma unroll
    for (int j = 0; j < 4; ++j) bX[j] = load_frag(wp + (size_t)j * 16 * K);
    v16h aY = load_frag(ap + 32);
    v16h bY[4];
    #pragma unroll
    for (int j = 0; j < 4; ++j) bY[j] = load_frag(wp + (size_t)j * 16 * K + 32);

    // ---- steady state ----------------------------------------------------
    for (int k0 = 64; k0 < K; k0 += 64) {
        // consume X (k0-64), refill X with k0
        #pragma unroll
        for (int j = 0; j < 4; ++j)
            acc[j] = __builtin_amdgcn_wmma_f32_16x16x32_f16(
                         false, aX, false, bX[j], (short)0, acc[j], false, false);
        aX = load_frag(ap + k0);
        #pragma unroll
        for (int j = 0; j < 4; ++j)
            bX[j] = load_frag(wp + (size_t)j * 16 * K + k0);

        __builtin_prefetch(ap + k0 + 256, 0, 1);   // global_prefetch_b8

        // consume Y (k0-32), refill Y with k0+32
        #pragma unroll
        for (int j = 0; j < 4; ++j)
            acc[j] = __builtin_amdgcn_wmma_f32_16x16x32_f16(
                         false, aY, false, bY[j], (short)0, acc[j], false, false);
        aY = load_frag(ap + k0 + 32);
        #pragma unroll
        for (int j = 0; j < 4; ++j)
            bY[j] = load_frag(wp + (size_t)j * 16 * K + k0 + 32);
    }

    // ---- epilogue: consume the final two buffer fills --------------------
    #pragma unroll
    for (int j = 0; j < 4; ++j)
        acc[j] = __builtin_amdgcn_wmma_f32_16x16x32_f16(
                     false, aX, false, bX[j], (short)0, acc[j], false, false);
    #pragma unroll
    for (int j = 0; j < 4; ++j)
        acc[j] = __builtin_amdgcn_wmma_f32_16x16x32_f16(
                     false, aY, false, bY[j], (short)0, acc[j], false, false);

    #pragma unroll
    for (int j = 0; j < 4; ++j) {
        #pragma unroll
        for (int v = 0; v < 8; ++v) {
            int m = m0 + v + 8 * kg;             // C layout: lane>=16 => M+8
            int n = n0 + j * 16 + row;
            float val = acc[j][v];
            if (silu) val = val / (1.f + __expf(-val));
            C[(size_t)m * N + n] = val;
        }
    }
}

// ---------------------------------------------------------------------------
// Sequential RWKV scan per head + GroupNorm(64) + gate.
// Block = 512 threads = 64 e-columns x 8 d-groups (8 d per thread, state in regs).
// y[e]     = sum_d r[d]*S[d][e]            (before update)
// c        = sum_d u[d]*r[d]*k[d]          (rank-1 diag term: out_diag = c*v[e])
// S[d][e]  = S[d][e]*w[d] + k[d]*v[e]
// ---------------------------------------------------------------------------
__global__ __launch_bounds__(512) void rwkv_scan_kernel(
        const float* __restrict__ r_f, const float* __restrict__ k_f,
        const float* __restrict__ v_f, const float* __restrict__ g_f,
        const float* __restrict__ time_decay, const float* __restrict__ time_faaaa,
        const float* __restrict__ ln_w, const float* __restrict__ ln_b,
        _Float16* __restrict__ a_out)
{
    const int h   = blockIdx.x;
    const int tid = threadIdx.x;
    const int e   = tid & 63;
    const int dg  = tid >> 6;        // 0..7
    const int d0  = dg * 8;

    __shared__ float rs[64], ks[64], vs[64];
    __shared__ float ysum[8 * 64];
    __shared__ float c8[8];
    __shared__ float os[64];
    __shared__ float lnw_s[64], lnb_s[64];
    __shared__ float stats[2];

    float S[8], wreg[8], ureg[8];
    #pragma unroll
    for (int i = 0; i < 8; ++i) {
        int d = d0 + i;
        S[i]    = 0.f;
        wreg[i] = __expf(-__expf(time_decay[h * 64 + d]));
        ureg[i] = time_faaaa[h * 64 + d];
    }
    if (tid < 64) { lnw_s[tid] = ln_w[h * 64 + tid]; lnb_s[tid] = ln_b[h * 64 + tid]; }
    __syncthreads();

    const size_t hbase = (size_t)h * 64;
    for (int t = 0; t < TLEN; ++t) {
        const size_t base = (size_t)t * HIDDEN + hbase;
        if      (tid < 64)  rs[tid]       = r_f[base + tid];
        else if (tid < 128) ks[tid - 64]  = k_f[base + tid - 64];
        else if (tid < 192) vs[tid - 128] = v_f[base + tid - 128];
        __syncthreads();

        float ve = vs[e];
        float yacc = 0.f;
        #pragma unroll
        for (int i = 0; i < 8; ++i) {
            int d = d0 + i;
            float rd = rs[d];
            yacc += rd * S[i];
            S[i] = S[i] * wreg[i] + ks[d] * ve;
        }
        ysum[dg * 64 + e] = yacc;
        if (e == 0) {
            float cp = 0.f;
            #pragma unroll
            for (int i = 0; i < 8; ++i) { int d = d0 + i; cp += ureg[i] * rs[d] * ks[d]; }
            c8[dg] = cp;
        }
        __syncthreads();

        float o8 = 0.f;
        if (tid < 64) {
            float cs = c8[0]+c8[1]+c8[2]+c8[3]+c8[4]+c8[5]+c8[6]+c8[7];
            float ys = 0.f;
            #pragma unroll
            for (int j = 0; j < 8; ++j) ys += ysum[j * 64 + e];
            o8 = (cs * vs[e] + ys) * 0.125f;     // / HEAD_SIZE_DIVISOR
            os[tid] = o8;
        }
        __syncthreads();
        if (tid < 32) {                          // GroupNorm stats, wave32 reduce
            float s = os[tid] + os[tid + 32];
            float q = os[tid]*os[tid] + os[tid + 32]*os[tid + 32];
            #pragma unroll
            for (int off = 16; off >= 1; off >>= 1) {
                s += __shfl_xor(s, off, 32);
                q += __shfl_xor(q, off, 32);
            }
            if (tid == 0) {
                float mu  = s * (1.f / 64.f);
                float var = q * (1.f / 64.f) - mu * mu;
                stats[0] = mu;
                stats[1] = rsqrtf(var + 1e-5f);
            }
        }
        __syncthreads();
        if (tid < 64) {
            float val = (o8 - stats[0]) * stats[1] * lnw_s[e] + lnb_s[e];
            a_out[base + e] = (_Float16)(val * g_f[base + e]);
        }
        __syncthreads();                         // protect LDS for next step
    }
}

// ---------------------------------------------------------------------------
extern "C" void kernel_launch(void* const* d_in, const int* in_sizes, int n_in,
                              void* d_out, int out_size, void* d_ws, size_t ws_size,
                              hipStream_t stream)
{
    const float* x    = (const float*)d_in[0];
    const int*   pos  = (const int*)  d_in[1];
    const float* W_r  = (const float*)d_in[2];
    const float* W_k  = (const float*)d_in[3];
    const float* W_v  = (const float*)d_in[4];
    const float* W_g  = (const float*)d_in[5];
    const float* W_o  = (const float*)d_in[6];
    const float* tm_k = (const float*)d_in[7];
    const float* tm_v = (const float*)d_in[8];
    const float* tm_r = (const float*)d_in[9];
    const float* tm_g = (const float*)d_in[10];
    const float* time_decay = (const float*)d_in[11];
    const float* time_faaaa = (const float*)d_in[12];
    const float* ln_w = (const float*)d_in[13];
    const float* ln_b = (const float*)d_in[14];
    float* out = (float*)d_out;

    const size_t NW = (size_t)HIDDEN * HIDDEN;   // 4M elems
    const size_t TC = (size_t)TLEN * HIDDEN;     // 4M elems

    char* ws = (char*)d_ws;
    _Float16* Wr_h = (_Float16*)ws; ws += NW * sizeof(_Float16);
    _Float16* Wk_h = (_Float16*)ws; ws += NW * sizeof(_Float16);
    _Float16* Wv_h = (_Float16*)ws; ws += NW * sizeof(_Float16);
    _Float16* Wg_h = (_Float16*)ws; ws += NW * sizeof(_Float16);
    _Float16* Wo_h = (_Float16*)ws; ws += NW * sizeof(_Float16);
    _Float16* xk_h = (_Float16*)ws; ws += TC * sizeof(_Float16);
    _Float16* xv_h = (_Float16*)ws; ws += TC * sizeof(_Float16);
    _Float16* xr_h = (_Float16*)ws; ws += TC * sizeof(_Float16);
    _Float16* xg_h = (_Float16*)ws; ws += TC * sizeof(_Float16);
    float*    k_f  = (float*)ws;    ws += TC * sizeof(float);
    float*    v_f  = (float*)ws;    ws += TC * sizeof(float);
    float*    r_f  = (float*)ws;    ws += TC * sizeof(float);
    float*    g_f  = (float*)ws;    ws += TC * sizeof(float);
    _Float16* a_h  = (_Float16*)ws; ws += TC * sizeof(_Float16);

    // 1) weight conversion (f32 -> f16)
    {
        dim3 blk(256), grd(8192);
        f32_to_f16_kernel<<<grd, blk, 0, stream>>>(W_r, Wr_h, (int)NW);
        f32_to_f16_kernel<<<grd, blk, 0, stream>>>(W_k, Wk_h, (int)NW);
        f32_to_f16_kernel<<<grd, blk, 0, stream>>>(W_v, Wv_h, (int)NW);
        f32_to_f16_kernel<<<grd, blk, 0, stream>>>(W_g, Wg_h, (int)NW);
        f32_to_f16_kernel<<<grd, blk, 0, stream>>>(W_o, Wo_h, (int)NW);
    }

    // 2) token-shift mixing
    mix_kernel<<<dim3((unsigned)(TC / 256)), dim3(256), 0, stream>>>(
        x, pos, tm_k, tm_v, tm_r, tm_g, xk_h, xv_h, xr_h, xg_h);

    // 3) projection GEMMs (k, v, r, silu-gate g)
    {
        dim3 blk(256), grd(HIDDEN / 64, TLEN / 128);
        wmma_gemm_f16<<<grd, blk, 0, stream>>>(xk_h, Wk_h, k_f, TLEN, HIDDEN, HIDDEN, 0);
        wmma_gemm_f16<<<grd, blk, 0, stream>>>(xv_h, Wv_h, v_f, TLEN, HIDDEN, HIDDEN, 0);
        wmma_gemm_f16<<<grd, blk, 0, stream>>>(xr_h, Wr_h, r_f, TLEN, HIDDEN, HIDDEN, 0);
        wmma_gemm_f16<<<grd, blk, 0, stream>>>(xg_h, Wg_h, g_f, TLEN, HIDDEN, HIDDEN, 1);
    }

    // 4) sequential scan + groupnorm + gate (f32 VALU)
    rwkv_scan_kernel<<<dim3(HEADS), dim3(512), 0, stream>>>(
        r_f, k_f, v_f, g_f, time_decay, time_faaaa, ln_w, ln_b, a_h);

    // 5) output projection
    {
        dim3 blk(256), grd(HIDDEN / 64, TLEN / 128);
        wmma_gemm_f16<<<grd, blk, 0, stream>>>(a_h, Wo_h, out, TLEN, HIDDEN, HIDDEN, 0);
    }
}
